// ProTran_42502996361295
// MI455X (gfx1250) — matrix operand order repaired and checked by reading, not compiled, not run on hardware
//
#include <hip/hip_runtime.h>

// ---------------------------------------------------------------------------
// ProTran forward for MI455X (gfx1250, wave32, WMMA).
//
// Strategy (see analysis): bf16 WMMA 16x16x32 GEMMs with f32 accumulation,
// incremental KV caches, GEMM epilogues fused with bias/residual/pos/LayerNorm
// or head-scatter into KV caches. Weights pre-packed into WMMA B-fragment
// layout so each lane loads its fragment as one contiguous 32B chunk.
// ---------------------------------------------------------------------------

#define B_   128
#define TS_  96
#define T_   120   // TS + FUT
#define Dm   512
#define Dl   256
#define H_   8
#define Lc   3

typedef __bf16 bf16_t;
typedef bf16_t v16bf __attribute__((ext_vector_type(16)));
typedef bf16_t v8bf  __attribute__((ext_vector_type(8)));
typedef float  v8f   __attribute__((ext_vector_type(8)));

__device__ __forceinline__ bf16_t f2bf(float f) {
  unsigned u = __float_as_uint(f);
  unsigned r = (u + 0x7FFFu + ((u >> 16) & 1u)) >> 16;   // round-to-nearest-even
  unsigned short s = (unsigned short)r;
  return __builtin_bit_cast(bf16_t, s);
}

// ---------------------------------------------------------------------------
// Pack fp32 weight [K,512] into WMMA B-fragment order:
//   packed[((nt*(K/32)+ks)*32 + lane)*16 + j] = bf16(W[k,n])
//   n = nt*16 + lane%16 ;  k = ks*32 + (lane/16)*16 + j
// (contiguous k order used here, same as the A loads: lanes 0-15 hold K=0..15, lanes 16-31 hold K=16..31)
// ---------------------------------------------------------------------------
__global__ __launch_bounds__(256)
void pack_w_kernel(const float* __restrict__ W, bf16_t* __restrict__ P, int K) {
  long idx  = (long)blockIdx.x * 256 + threadIdx.x;    // total K*512
  int  j    = (int)(idx & 15);
  int  lane = (int)((idx >> 4) & 31);
  long t2   = idx >> 9;
  int  Ks   = K >> 5;
  int  ks   = (int)(t2 % Ks);
  int  nt   = (int)(t2 / Ks);
  int  half = lane >> 4;
  int  n    = nt * 16 + (lane & 15);
  int  k    = ks * 32 + half * 16 + j;
  P[idx] = f2bf(W[(long)k * Dm + n]);
}

__global__ __launch_bounds__(256)
void cvt_bf_kernel(const float* __restrict__ s, bf16_t* __restrict__ d, long n) {
  long i = (long)blockIdx.x * 256 + threadIdx.x;
  if (i < n) d[i] = f2bf(s[i]);
}

// gather one 512-wide f32 row (arbitrary stride) -> compact bf16 [rows,512]
__global__ __launch_bounds__(256)
void gather_rows_kernel(const float* __restrict__ src, long ld, bf16_t* __restrict__ dst) {
  long r = blockIdx.x; int t = threadIdx.x;
  dst[r * Dm + t]       = f2bf(src[r * ld + t]);
  dst[r * Dm + 256 + t] = f2bf(src[r * ld + 256 + t]);
}

// W_prev rows 1..T of [B,T+1,512] -> compact bf16 [B*T,512]
__global__ __launch_bounds__(256)
void gather_wprev_kernel(const float* __restrict__ w, bf16_t* __restrict__ dst) {
  long r = blockIdx.x;                 // 0..B*T-1
  long b = r / T_, t = r % T_;
  const float* s = w + (b * (T_ + 1) + 1 + t) * (long)Dm;
  int x = threadIdx.x;
  dst[r * Dm + x]       = f2bf(s[x]);
  dst[r * Dm + 256 + x] = f2bf(s[256 + x]);
}

__global__ __launch_bounds__(256)
void init_w_kernel(const float* __restrict__ w0, float* __restrict__ w) {
  long b = blockIdx.x; int t = threadIdx.x;
  w[b * (long)(T_ + 1) * Dm + t]       = w0[t];
  w[b * (long)(T_ + 1) * Dm + 256 + t] = w0[256 + t];
}

// ---------------------------------------------------------------------------
// GEMM  C[M,512] = A[M,K](bf16) @ Wpk[K,512](packed bf16) + bias, epilogues:
//   mode 0: plain f32 store (ldc) + optional bf16 copy (stride 512)
//   mode 1: LayerNorm( acc + bias + res + pos ) * g + be  (full row in WG)
//   mode 2: head-scatter into KV cache [B,H,cap,64]
// Workgroup: 256 threads / 8 waves; wave w -> 16 rows x cols [64w,64w+64).
// ---------------------------------------------------------------------------
__global__ __launch_bounds__(256)
void gemm512_kernel(const bf16_t* __restrict__ A, long lda,
                    const bf16_t* __restrict__ Wpk, int K,
                    const float* __restrict__ bias,
                    int mode,
                    float* __restrict__ C, long ldc,
                    bf16_t* __restrict__ Cbf,
                    const float* __restrict__ res, long res_ld,
                    const float* __restrict__ pos, int pos_mod,
                    const float* __restrict__ g, const float* __restrict__ be,
                    int rows_per_b, int t_off, int cap) {
  __shared__ float lds[16 * 512];
  const int wave = threadIdx.x >> 5;
  const int lane = threadIdx.x & 31;
  const int half = lane >> 4;
  const int mr   = lane & 15;
  const long m0  = (long)blockIdx.x * 16;
  const int Ks   = K >> 5;

  v8f acc[4] = {};
  const bf16_t* Arow = A + (m0 + mr) * lda;
  for (int ks = 0; ks < Ks; ++ks) {
    const int kb = ks * 32;
    v16bf av = *(const v16bf*)(Arow + kb + half * 16);
#pragma unroll
    for (int q = 0; q < 4; ++q) {
      const int nt = wave * 4 + q;
      v16bf bv = *(const v16bf*)(Wpk + (((long)nt * Ks + ks) * 32 + lane) * 16);
      acc[q] = __builtin_amdgcn_wmma_f32_16x16x32_bf16(
          false, av, false, bv, (short)0, acc[q], false, false);
    }
  }

  // D layout: lane holds col n = nt*16 + lane%16; element r -> row r + 8*(lane/16)
#pragma unroll
  for (int q = 0; q < 4; ++q) {
    const int n = wave * 64 + q * 16 + (lane & 15);
    const float bsum = bias ? bias[n] : 0.0f;
#pragma unroll
    for (int r = 0; r < 8; ++r) {
      float v = acc[q][r] + bsum;
      const int lm = r + 8 * half;
      const long gm = m0 + lm;
      if (mode == 2) {
        long b = gm / rows_per_b;
        int  tt = t_off + (int)(gm % rows_per_b);
        int  h = n >> 6, d = n & 63;
        C[((b * H_ + h) * (long)cap + tt) * 64 + d] = v;
      } else if (mode == 0) {
        C[gm * ldc + n] = v;
        if (Cbf) Cbf[gm * (long)Dm + n] = f2bf(v);
      } else {
        lds[lm * 512 + n] = v;
      }
    }
  }

  if (mode == 1) {
    __syncthreads();
    for (int rr = 0; rr < 2; ++rr) {
      const int  lm = wave * 2 + rr;
      const long gm = m0 + lm;
      const float* rrow = res ? res + gm * res_ld : nullptr;
      const float* prow = pos ? pos + (pos_mod > 0 ? (gm % pos_mod) * (long)Dm : 0) : nullptr;
      float x[16]; float s1 = 0.f, s2 = 0.f;
#pragma unroll
      for (int s = 0; s < 16; ++s) {
        const int d = lane + s * 32;
        float v = lds[lm * 512 + d];
        if (rrow) v += rrow[d];
        if (prow) v += prow[d];
        x[s] = v; s1 += v; s2 += v * v;
      }
      for (int off = 16; off; off >>= 1) {
        s1 += __shfl_xor(s1, off, 32);
        s2 += __shfl_xor(s2, off, 32);
      }
      const float mean = s1 * (1.0f / 512.0f);
      const float var  = s2 * (1.0f / 512.0f) - mean * mean;
      const float rstd = __frsqrt_rn(var + 1e-5f);
#pragma unroll
      for (int s = 0; s < 16; ++s) {
        const int d = lane + s * 32;
        const float y = (x[s] - mean) * rstd * g[d] + be[d];
        if (C)   C[gm * ldc + d] = y;
        if (Cbf) Cbf[gm * (long)Dm + d] = f2bf(y);
      }
    }
  }
}

// ---------------------------------------------------------------------------
// Single-query attention per (batch, head): q[64] vs K,V [cap,64], ctx rows.
// softmax in f32; output written bf16 head-major [B,512] (A of o-projection).
// ---------------------------------------------------------------------------
__global__ __launch_bounds__(128)
void attn_kernel(const float* __restrict__ q,
                 const float* __restrict__ K, const float* __restrict__ V,
                 int ctx, int cap, bf16_t* __restrict__ obf) {
  __shared__ float sp[128];
  __shared__ float sr[128];
  const long bh = blockIdx.x;
  const float* qv = q + (bh >> 3) * (long)Dm + (bh & 7) * 64;
  const float* Kb = K + bh * (long)cap * 64;
  const float* Vb = V + bh * (long)cap * 64;
  const int t = threadIdx.x;

  float sc = -3.0e38f;
  if (t < ctx) {
    float a = 0.f;
#pragma unroll 8
    for (int d = 0; d < 64; ++d) a += qv[d] * Kb[t * 64 + d];
    sc = a * 0.125f;                       // 1/sqrt(64)
  }
  sr[t] = sc; __syncthreads();
  for (int st = 64; st > 0; st >>= 1) { if (t < st) sr[t] = fmaxf(sr[t], sr[t + st]); __syncthreads(); }
  const float mx = sr[0]; __syncthreads();
  const float p = (t < ctx) ? __expf(sc - mx) : 0.f;
  sp[t] = p; sr[t] = p; __syncthreads();
  for (int st = 64; st > 0; st >>= 1) { if (t < st) sr[t] += sr[t + st]; __syncthreads(); }
  const float inv = 1.0f / sr[0];
  if (t < 64) {
    float a = 0.f;
    for (int j = 0; j < ctx; ++j) a += sp[j] * Vb[j * 64 + t];
    obf[(bh >> 3) * (long)Dm + (bh & 7) * 64 + t] = f2bf(a * inv);
  }
}

// z = mean + exp(0.5*logvar)*eps ; write z/mean/logvar to outputs, z->bf16
__global__ __launch_bounds__(256)
void sample_kernel(const float* __restrict__ stats, const float* __restrict__ eps_l, int i,
                   float* __restrict__ oz, float* __restrict__ om, float* __restrict__ olv,
                   bf16_t* __restrict__ zbf) {
  const long b = blockIdx.x; const int d = threadIdx.x;
  const float mean = stats[b * Dm + d];
  const float lv   = stats[b * Dm + 256 + d];
  const float e    = eps_l[(b * T_ + i) * (long)Dl + d];
  const float z    = mean + __expf(0.5f * lv) * e;
  const long off = (b * T_ + i) * (long)Dl + d;
  om[off] = mean; olv[off] = lv; oz[off] = z;
  zbf[b * Dl + d] = f2bf(z);
}

// xhat[b,t,p] = sum_l qz[b,t,l]*decW[t,p,l] + decb[t,p]
__global__ __launch_bounds__(64)
void decode_kernel(const float* __restrict__ qz, const float* __restrict__ decW,
                   const float* __restrict__ decb, float* __restrict__ xhat) {
  const long bt = blockIdx.x;                 // b*T + t
  const int t = (int)(bt % T_);
  const int p = threadIdx.x;
  const float* wr = decW + ((long)t * 64 + p) * Dl;
  const float* zr = qz + bt * (long)Dl;
  float a = decb[t * 64 + p];
  for (int d = 0; d < Dl; ++d) a += zr[d] * wr[d];
  xhat[bt * 64 + p] = a;
}

// ---------------------------------------------------------------------------
// Host orchestration
// ---------------------------------------------------------------------------
// Assumed flat input order (setup_inputs dict order, params pytree in dict-
// insertion DFS order; each mha = Wq,bq,Wk,bk,Wv,bv,Wo,bo):
//  0 ctx  1 tgt  2 eps_prior  3 eps_post  4 prior.w0
//  5..12 mha1  13..20 mha2  21..28 mha3
//  29 fc1_W 30 fc1_b 31 fc3_W 32 fc3_b
//  33..38 ln1_g,ln1_b,ln2_g,ln2_b,ln3_g,ln3_b
//  39 pos 40 posln_g 41 posln_b
//  42..49 post.mha  50 post.fc_W 51 post.fc_b 52 post.ln_g 53 post.ln_b
//  54 fcC_W 55 fcC_b 56 fcT_W 57 fcT_b
//  58 posC 59 posC_ln_g 60 posC_ln_b  61 posT 62 posT_ln_g 63 posT_ln_b
//  64 dec_W 65 dec_b

extern "C" void kernel_launch(void* const* d_in, const int* in_sizes, int n_in,
                              void* d_out, int out_size, void* d_ws, size_t ws_size,
                              hipStream_t stream) {
  (void)in_sizes; (void)out_size;
  if (n_in < 66) return;
  auto F = [&](int i) { return (const float*)d_in[i]; };

  // ---- workspace allocator ----
  char* base = (char*)d_ws; size_t off = 0;
  auto alloc = [&](size_t bytes) { void* p = base + off; off = (off + bytes + 255) & ~(size_t)255; return p; };
  auto af  = [&](size_t n) { return (float*)alloc(n * 4); };
  auto ab  = [&](size_t n) { return (bf16_t*)alloc(n * 2); };

  // packed weights
  struct WDef { int in; int K; };
  const WDef wd[21] = {
    {5,512},{7,512},{9,512},{11,512},      // mha1 Wq,Wk,Wv,Wo
    {13,512},{15,512},{17,512},{19,512},   // mha2
    {21,512},{23,512},{25,512},{27,512},   // mha3
    {42,512},{44,512},{46,512},{48,512},   // post mha
    {29,512},{50,512},{31,256},{54,64},{56,64} // fc1, post fc, fc3, fcC, fcT
  };
  bf16_t* pk[21];
  for (int i = 0; i < 21; ++i) pk[i] = ab((size_t)wd[i].K * Dm);

  bf16_t* ctxbf = ab((size_t)B_ * TS_ * 64);
  bf16_t* tgtbf = ab((size_t)B_ * T_ * 64);
  bf16_t* hCbf  = ab((size_t)B_ * TS_ * Dm);
  bf16_t* hTbf  = ab((size_t)B_ * T_ * Dm);
  bf16_t* stagebf = ab((size_t)B_ * T_ * Dm);
  bf16_t* qbf = ab((size_t)B_ * Dm);
  bf16_t* wtbf = ab((size_t)B_ * Dm);
  bf16_t* wbarbf = ab((size_t)B_ * Dm);
  bf16_t* whatbf = ab((size_t)B_ * Dm);
  bf16_t* k2bf = ab((size_t)B_ * Dm);
  bf16_t* attbf = ab((size_t)B_ * Dm);
  bf16_t* zbf = ab((size_t)B_ * Dl);

  float* wbuf4[4]; for (int i = 0; i < 4; ++i) wbuf4[i] = af((size_t)B_ * (T_ + 1) * Dm);
  float* K1 = af((size_t)B_ * H_ * T_ * 64);  float* V1 = af((size_t)B_ * H_ * T_ * 64);
  float* K3 = af((size_t)B_ * H_ * T_ * 64);  float* V3 = af((size_t)B_ * H_ * T_ * 64);
  float* K2 = af((size_t)B_ * H_ * TS_ * 64); float* V2 = af((size_t)B_ * H_ * TS_ * 64);
  float* K4 = af((size_t)B_ * H_ * T_ * 64);  float* V4 = af((size_t)B_ * H_ * T_ * 64);
  float* tq = af((size_t)B_ * Dm);
  float* wtf = af((size_t)B_ * Dm);
  float* wbarf = af((size_t)B_ * Dm);
  float* whatf = af((size_t)B_ * Dm);
  float* k2f = af((size_t)B_ * Dm);
  float* stats = af((size_t)B_ * Dm);
  if (off > ws_size) return;   // insufficient scratch (deterministic no-op)

  // ---- output slices ----
  float* out = (float*)d_out;
  const long szX = (long)B_ * T_ * 64;
  const long szZ = (long)B_ * T_ * Dl;
  float* o_xhat = out;
  float* o_pz = out + szX;
  float* o_pm = o_pz + Lc * szZ;
  float* o_plv = o_pm + Lc * szZ;
  float* o_qz = o_plv + Lc * szZ;
  float* o_qm = o_qz + Lc * szZ;
  float* o_qlv = o_qm + Lc * szZ;

  // ---- pack weights, convert inputs ----
  for (int i = 0; i < 21; ++i)
    pack_w_kernel<<<(wd[i].K * Dm) / 256, 256, 0, stream>>>(F(wd[i].in), pk[i], wd[i].K);
  cvt_bf_kernel<<<(int)(((long)B_ * TS_ * 64 + 255) / 256), 256, 0, stream>>>(F(0), ctxbf, (long)B_ * TS_ * 64);
  cvt_bf_kernel<<<(int)(((long)B_ * T_ * 64 + 255) / 256), 256, 0, stream>>>(F(1), tgtbf, (long)B_ * T_ * 64);

  auto gemm = [&](int M, const bf16_t* A, long lda, const bf16_t* Wp, int K, const float* bias,
                  int mode, float* C, long ldc, bf16_t* Cbf,
                  const float* res, long res_ld, const float* pos, int pos_mod,
                  const float* g, const float* be, int rpb, int t_off, int cap) {
    gemm512_kernel<<<M / 16, 256, 0, stream>>>(A, lda, Wp, K, bias, mode, C, ldc, Cbf,
                                               res, res_ld, pos, pos_mod, g, be, rpb, t_off, cap);
  };

  // ---- embeddings: h_C = LN(ctx@fcC + b + posC), h_T likewise (bf16 only) ----
  gemm(B_ * TS_, ctxbf, 64, pk[19], 64, F(55), 1, nullptr, 0, hCbf,
       nullptr, 0, F(58), TS_, F(59), F(60), 0, 0, 0);
  gemm(B_ * T_, tgtbf, 64, pk[20], 64, F(57), 1, nullptr, 0, hTbf,
       nullptr, 0, F(61), T_, F(62), F(63), 0, 0, 0);

  // ---- constant cross-attention KV caches ----
  gemm(B_ * TS_, hCbf, Dm, pk[5], Dm, F(15 + 1), 2, K2, 0, nullptr, nullptr, 0, nullptr, 0, nullptr, nullptr, TS_, 0, TS_); // mha2.Wk,bk (idx16)
  gemm(B_ * TS_, hCbf, Dm, pk[6], Dm, F(17 + 1), 2, V2, 0, nullptr, nullptr, 0, nullptr, 0, nullptr, nullptr, TS_, 0, TS_); // mha2.Wv,bv
  gemm(B_ * T_, hTbf, Dm, pk[13], Dm, F(45), 2, K4, 0, nullptr, nullptr, 0, nullptr, 0, nullptr, nullptr, T_, 0, T_);       // post.Wk,bk
  gemm(B_ * T_, hTbf, Dm, pk[14], Dm, F(47), 2, V4, 0, nullptr, nullptr, 0, nullptr, 0, nullptr, nullptr, T_, 0, T_);       // post.Wv,bv

  const float* w0 = F(4);
  const float *bq1=F(6), *bk1=F(8), *bv1=F(10), *bo1=F(12);
  const float *bq2=F(14), *bo2=F(20);
  const float *bq3=F(22), *bk3=F(24), *bv3=F(26), *bo3=F(28);
  const float *bq4=F(43), *bo4=F(49);
  const float *fc1b=F(30), *fc3b=F(32), *fcPb=F(51);
  const float *ln1g=F(33), *ln1b=F(34), *ln2g=F(35), *ln2b=F(36), *ln3g=F(37), *ln3b=F(38);
  const float *ppos=F(39), *plng=F(40), *plnb=F(41);
  const float *png=F(52), *pnb=F(53);
  const long wld = (long)(T_ + 1) * Dm;

  auto run_module = [&](bool is_post, int l, const float* w_prev, float* wbuf,
                        const float* eps_l, float* oz, float* om, float* olv) {
    if (l > 0) {
      gather_wprev_kernel<<<B_ * T_, 256, 0, stream>>>(w_prev, stagebf);
      gemm(B_ * T_, stagebf, Dm, pk[9], Dm, bk3, 2, K3, 0, nullptr, nullptr, 0, nullptr, 0, nullptr, nullptr, T_, 0, T_);
      gemm(B_ * T_, stagebf, Dm, pk[10], Dm, bv3, 2, V3, 0, nullptr, nullptr, 0, nullptr, 0, nullptr, nullptr, T_, 0, T_);
    }
    init_w_kernel<<<B_, 256, 0, stream>>>(w0, wbuf);

    for (int i = 0; i < T_; ++i) {
      const float* qrow = wbuf + (long)i * Dm;
      gather_rows_kernel<<<B_, 256, 0, stream>>>(qrow, wld, qbf);

      const bf16_t* wt_bf; const float* wt_res; long wt_ld;
      if (l > 0) {  // w_t = LN(q + MHA3(q, W_prev))
        gemm(B_, qbf, Dm, pk[8], Dm, bq3, 0, tq, Dm, nullptr, nullptr, 0, nullptr, 0, nullptr, nullptr, 0, 0, 0);
        attn_kernel<<<B_ * H_, 128, 0, stream>>>(tq, K3, V3, T_, T_, attbf);
        gemm(B_, attbf, Dm, pk[11], Dm, bo3, 1, wtf, Dm, wtbf, qrow, wld, nullptr, 0, ln3g, ln3b, 0, 0, 0);
        wt_bf = wtbf; wt_res = wtf; wt_ld = Dm;
      } else { wt_bf = qbf; wt_res = qrow; wt_ld = wld; }

      // mha1 self-attention: append KV from original q, query from w_t
      gemm(B_, qbf, Dm, pk[1], Dm, bk1, 2, K1, 0, nullptr, nullptr, 0, nullptr, 0, nullptr, nullptr, 1, i, T_);
      gemm(B_, qbf, Dm, pk[2], Dm, bv1, 2, V1, 0, nullptr, nullptr, 0, nullptr, 0, nullptr, nullptr, 1, i, T_);
      gemm(B_, wt_bf, Dm, pk[0], Dm, bq1, 0, tq, Dm, nullptr, nullptr, 0, nullptr, 0, nullptr, nullptr, 0, 0, 0);
      attn_kernel<<<B_ * H_, 128, 0, stream>>>(tq, K1, V1, i + 1, T_, attbf);
      gemm(B_, attbf, Dm, pk[3], Dm, bo1, 1, wbarf, Dm, wbarbf, wt_res, wt_ld, nullptr, 0, ln1g, ln1b, 0, 0, 0);

      // mha2 cross-attn over h_C
      gemm(B_, wbarbf, Dm, pk[4], Dm, bq2, 0, tq, Dm, nullptr, nullptr, 0, nullptr, 0, nullptr, nullptr, 0, 0, 0);
      attn_kernel<<<B_ * H_, 128, 0, stream>>>(tq, K2, V2, TS_, TS_, attbf);
      gemm(B_, attbf, Dm, pk[7], Dm, bo2, 1, whatf, Dm, whatbf, wbarf, Dm, nullptr, 0, ln2g, ln2b, 0, 0, 0);

      if (is_post) {
        gemm(B_, whatbf, Dm, pk[12], Dm, bq4, 0, tq, Dm, nullptr, nullptr, 0, nullptr, 0, nullptr, nullptr, 0, 0, 0);
        attn_kernel<<<B_ * H_, 128, 0, stream>>>(tq, K4, V4, T_, T_, attbf);
        gemm(B_, attbf, Dm, pk[15], Dm, bo4, 1, k2f, Dm, k2bf, whatf, Dm, nullptr, 0, png, pnb, 0, 0, 0);
        gemm(B_, k2bf, Dm, pk[17], Dm, fcPb, 0, stats, Dm, nullptr, nullptr, 0, nullptr, 0, nullptr, nullptr, 0, 0, 0);
      } else {
        gemm(B_, whatbf, Dm, pk[16], Dm, fc1b, 0, stats, Dm, nullptr, nullptr, 0, nullptr, 0, nullptr, nullptr, 0, 0, 0);
      }

      sample_kernel<<<B_, 256, 0, stream>>>(stats, eps_l, i, oz, om, olv, zbf);

      // w_new = LN(w_hat + z@fc3 + b + pos[i]) -> w row i+1
      gemm(B_, zbf, Dl, pk[18], Dl, fc3b, 1, wbuf + (long)(i + 1) * Dm, wld, nullptr,
           whatf, Dm, ppos + (long)i * Dm, 0, plng, plnb, 0, 0, 0);
    }
  };

  const float* pPrev = nullptr; const float* qPrev = nullptr;
  for (int l = 0; l < Lc; ++l) {
    float* pCur = wbuf4[0 + (l & 1)];
    float* qCur = wbuf4[2 + (l & 1)];
    run_module(false, l, pPrev, pCur, F(2) + (long)l * szZ, o_pz + l * szZ, o_pm + l * szZ, o_plv + l * szZ);
    run_module(true,  l, qPrev, qCur, F(3) + (long)l * szZ, o_qz + l * szZ, o_qm + l * szZ, o_qlv + l * szZ);
    pPrev = pCur; qPrev = qCur;
  }

  decode_kernel<<<B_ * T_, 64, 0, stream>>>(o_qz + (Lc - 1) * szZ, F(64), F(65), o_xhat);
}
